// ratAttSublayer_36644660970135
// MI455X (gfx1250) — matrix operand order repaired and checked by reading, not compiled, not run on hardware
//
#include <hip/hip_runtime.h>
#include <hip/hip_bf16.h>

// Problem constants (match reference)
#define Bc  2
#define Sc  1024
#define Dc  1024
#define Hc  16
#define DKc 64
#define NRc 16

typedef __attribute__((ext_vector_type(16))) __bf16 v16bf;
typedef __attribute__((ext_vector_type(8)))  __bf16 v8bf;
typedef __attribute__((ext_vector_type(4)))  __bf16 v4bf;
typedef __attribute__((ext_vector_type(8)))  float  v8f;
typedef unsigned int v4u  __attribute__((ext_vector_type(4)));
typedef int          v8i_ __attribute__((ext_vector_type(8)));
typedef int          v4i_ __attribute__((ext_vector_type(4)));

// LDS A-panel: 16 rows x 1024 bf16, +4 elements pad every 512 elements
// (matches TDM pad_interval=256 DWORDs / pad_amount=2 DWORDs) -> each row is
// shifted by 4 banks, de-conflicting the 16-lane halves of ds_load_b128.
#define A_LDS_ELEMS (16 * Dc + (16 * Dc / 512) * 4)   // 16512
__device__ __forceinline__ int a_lds_off(int linear) {
    return linear + ((linear >> 9) << 2);
}

__device__ __forceinline__ v8f wmma_bf16(v16bf a, v16bf b, v8f c) {
    return __builtin_amdgcn_wmma_f32_16x16x32_bf16(
        false, a, false, b, (short)0, c, false, false);
}

// ---- CDNA5 WMMA fragment loaders (ISA 7.12.2, wave32), b128 vectorized ----
__device__ __forceinline__ v16bf load_a_bf(const __bf16* __restrict__ src,
                                           int lda, int lane) {
    int row = lane & 15;
    int kb  = (lane & 16) >> 1;                  // 0 or 8
    const __bf16* base = src + row * lda + kb;
    v8bf lo = *(const v8bf*)(base);
    v8bf hi = *(const v8bf*)(base + 16);
    v16bf a;
#pragma unroll
    for (int i = 0; i < 8; ++i) { a[i] = lo[i]; a[8 + i] = hi[i]; }
    return a;
}

// A fragment from the padded LDS A panel (row stride Dc + pad map)
__device__ __forceinline__ v16bf load_a_As(const __bf16* As, int k, int lane) {
    int row = lane & 15;
    int kb  = (lane & 16) >> 1;
    int o0 = row * Dc + k + kb;                  // 8-element run (multiple of 8)
    int o1 = o0 + 16;
    v8bf lo = *(const v8bf*)(As + a_lds_off(o0));
    v8bf hi = *(const v8bf*)(As + a_lds_off(o1));
    v16bf a;
#pragma unroll
    for (int i = 0; i < 8; ++i) { a[i] = lo[i]; a[8 + i] = hi[i]; }
    return a;
}

__device__ __forceinline__ v16bf load_b_nmajor(const __bf16* __restrict__ src,
                                               int ldn, int lane) {
    int col = lane & 15;
    int kb  = (lane & 16) >> 1;                  // 0 or 8
    const __bf16* base = src + col * ldn + kb;
    v8bf lo = *(const v8bf*)(base);
    v8bf hi = *(const v8bf*)(base + 16);
    v16bf b;
#pragma unroll
    for (int i = 0; i < 8; ++i) { b[i] = lo[i]; b[8 + i] = hi[i]; }
    return b;
}

__device__ __forceinline__ v16bf load_a_lds_f32(const float* __restrict__ src,
                                                int lda, int lane) {
    int row = lane & 15;
    int kb  = (lane & 16) >> 1;
    const float4* p0 = (const float4*)(src + row * lda + kb);
    const float4* p1 = (const float4*)(src + row * lda + kb + 16);
    float4 x0 = p0[0], x1 = p0[1];
    float4 y0 = p1[0], y1 = p1[1];
    v16bf a;
    a[0] = (__bf16)x0.x; a[1] = (__bf16)x0.y; a[2] = (__bf16)x0.z; a[3] = (__bf16)x0.w;
    a[4] = (__bf16)x1.x; a[5] = (__bf16)x1.y; a[6] = (__bf16)x1.z; a[7] = (__bf16)x1.w;
    a[8] = (__bf16)y0.x; a[9] = (__bf16)y0.y; a[10] = (__bf16)y0.z; a[11] = (__bf16)y0.w;
    a[12] = (__bf16)y1.x; a[13] = (__bf16)y1.y; a[14] = (__bf16)y1.z; a[15] = (__bf16)y1.w;
    return a;
}

// ---- TDM: stage a 16xDc bf16 panel into (padded) LDS ----------------------
// D# per CDNA5 ISA ch.8: group0 = {flags, lds_addr, global_addr, type=2},
// group1 = {data_size=2B, pad(256dw interval, 2dw amount), dims/strides}.
// Toolchain uses the 6-arg builtin: (g0, g1, g2, g3, g4, cpol).
__device__ __forceinline__ void stage_a_panel(__bf16* As,
                                              const __bf16* __restrict__ Xrow,
                                              int tid) {
#if __has_builtin(__builtin_amdgcn_tensor_load_to_lds) && \
    __has_builtin(__builtin_amdgcn_s_wait_tensorcnt)
    if (tid < 32) {                      // one wave issues; TDM ignores EXEC
        unsigned long long ga = (unsigned long long)(uintptr_t)Xrow;
        v4u g0;
        g0[0] = 1u;                                       // count=1 valid D#
        g0[1] = (unsigned)(uintptr_t)As;                  // LDS byte address
        g0[2] = (unsigned)(ga & 0xFFFFFFFFu);             // global_addr[31:0]
        g0[3] = (unsigned)((ga >> 32) & 0x01FFFFFFu) | (2u << 30);  // type=2
        v8i_ g1;
        g1[0] = (1 << 16) | (1 << 20) | (7 << 22) | (1 << 25);
        //        data2B    pad_en      intv=256dw  amount=2dw
        g1[1] = (int)((Dc & 0xFFFF) << 16);               // tensor_dim0 lo
        g1[2] = (int)((Dc >> 16) | ((16 & 0xFFFF) << 16));// dim0 hi | dim1 lo
        g1[3] = (int)((16 >> 16) | (Dc << 16));           // dim1 hi | tile_dim0
        g1[4] = 16;                                       // tile_dim1 (dim2=0)
        g1[5] = Dc;                                       // dim0_stride[31:0]
        g1[6] = 0;
        g1[7] = 0;
        v4i_ z4 = {0, 0, 0, 0};
        v8i_ z8 = {0, 0, 0, 0, 0, 0, 0, 0};
        __builtin_amdgcn_tensor_load_to_lds(g0, g1, z4, z4, z8, 0);
        __builtin_amdgcn_s_wait_tensorcnt(0);
    }
    __syncthreads();
#else
    // cooperative fallback: identical padded layout
    for (int o = tid * 8; o < 16 * Dc; o += 256 * 8) {
        v8bf v = *(const v8bf*)(Xrow + o);                // panel is contiguous
        *(v8bf*)(As + a_lds_off(o)) = v;
    }
    __syncthreads();
#endif
}

// ---- software-pipelined 16x64 GEMM main loop (A from LDS, B from global) --
__device__ __forceinline__ void gemm_mainloop(const __bf16* As,
                                              const __bf16* __restrict__ Bbase,
                                              int lane, v8f acc[4]) {
    v16bf a  = load_a_As(As, 0, lane);
    v16bf b0 = load_b_nmajor(Bbase, Dc, lane);
    v16bf b1 = load_b_nmajor(Bbase + 16 * Dc, Dc, lane);
    v16bf b2 = load_b_nmajor(Bbase + 32 * Dc, Dc, lane);
    v16bf b3 = load_b_nmajor(Bbase + 48 * Dc, Dc, lane);
    for (int k = 32; k < Dc; k += 32) {
        __builtin_prefetch(Bbase + k + 32, 0, 3);         // global_prefetch_b8
        v16bf an = load_a_As(As, k, lane);
        v16bf c0 = load_b_nmajor(Bbase + k, Dc, lane);
        v16bf c1 = load_b_nmajor(Bbase + 16 * Dc + k, Dc, lane);
        v16bf c2 = load_b_nmajor(Bbase + 32 * Dc + k, Dc, lane);
        v16bf c3 = load_b_nmajor(Bbase + 48 * Dc + k, Dc, lane);
        acc[0] = wmma_bf16(a, b0, acc[0]);
        acc[1] = wmma_bf16(a, b1, acc[1]);
        acc[2] = wmma_bf16(a, b2, acc[2]);
        acc[3] = wmma_bf16(a, b3, acc[3]);
        a = an; b0 = c0; b1 = c1; b2 = c2; b3 = c3;
    }
    acc[0] = wmma_bf16(a, b0, acc[0]);
    acc[1] = wmma_bf16(a, b1, acc[1]);
    acc[2] = wmma_bf16(a, b2, acc[2]);
    acc[3] = wmma_bf16(a, b3, acc[3]);
}

// ---- K0a: fp32 -> bf16 conversion -----------------------------------------
__global__ void cvt_bf16_kernel(const float* __restrict__ in,
                                __bf16* __restrict__ out, int n) {
    int i = (blockIdx.x * blockDim.x + threadIdx.x) * 4;
    if (i >= n) return;
    float4 v = *(const float4*)(in + i);
    v4bf o;
    o[0] = (__bf16)v.x; o[1] = (__bf16)v.y; o[2] = (__bf16)v.z; o[3] = (__bf16)v.w;
    *(v4bf*)(out + i) = o;
}

// ---- K0b: fp32 [K][N] -> bf16 [N][K] transpose-convert (weights) ----------
__global__ __launch_bounds__(256) void cvtT_bf16_kernel(
    const float* __restrict__ in, __bf16* __restrict__ out) {
    __shared__ float t[32][33];
    int n0 = blockIdx.x * 32;
    int k0 = blockIdx.y * 32;
    int tx = threadIdx.x & 31, ty = threadIdx.x >> 5;
#pragma unroll
    for (int r = 0; r < 32; r += 8)
        t[ty + r][tx] = in[(size_t)(k0 + ty + r) * Dc + n0 + tx];
    __syncthreads();
#pragma unroll
    for (int r = 0; r < 32; r += 8)
        out[(size_t)(n0 + ty + r) * Dc + k0 + tx] = (__bf16)t[tx][ty + r];
}

// ---- K1: projection GEMM; TDM-staged A panel shared by all 8 waves --------
// Block = one 16-row m-tile x 512 columns (8 waves x 64 cols).
// transpose_out==0: Out[b,h,s,d] (Q,K); ==1: Out[b,h,d,s] (V).
__global__ __launch_bounds__(256) void gemm_proj_kernel(
    const __bf16* __restrict__ X, const __bf16* __restrict__ WT,
    __bf16* __restrict__ Out, int transpose_out) {
    __shared__ __bf16 As[A_LDS_ELEMS];                   // 33 KB A panel
    __shared__ float ctile[8][16][64];                   // 32 KB epilogue
    int wave = threadIdx.x >> 5;
    int lane = threadIdx.x & 31;
    int mt = blockIdx.x >> 1;                            // shared by block
    int nb = ((blockIdx.x & 1) << 3) + wave;
    int m0 = mt * 16, n0 = nb * 64;

    stage_a_panel(As, X + (size_t)m0 * Dc, threadIdx.x);

    v8f acc[4] = {v8f{}, v8f{}, v8f{}, v8f{}};
    gemm_mainloop(As, WT + (size_t)n0 * Dc, lane, acc);

    int col   = lane & 15;
    int rbase = (lane & 16) >> 1;
    if (transpose_out) {
        int m = m0 + rbase;
        int bb = m >> 10, s = m & (Sc - 1);
#pragma unroll
        for (int t = 0; t < 4; ++t) {
            int n = n0 + t * 16 + col;
            int h = n >> 6, d = n & (DKc - 1);
            v8bf o;
#pragma unroll
            for (int e = 0; e < 8; ++e) o[e] = (__bf16)acc[t][e];
            *(v8bf*)(Out + ((size_t)(bb * Hc + h) * DKc + d) * Sc + s) = o;
        }
    } else {
        float (*ct)[64] = ctile[wave];
#pragma unroll
        for (int t = 0; t < 4; ++t)
#pragma unroll
            for (int e = 0; e < 8; ++e)
                ct[rbase + e][t * 16 + col] = acc[t][e];
        int row = lane & 15, ch = (lane >> 4) * 32;
        int m = m0 + row;
        int bb = m >> 10, s = m & (Sc - 1);
        int h = n0 >> 6;
        __bf16* dst = Out + ((size_t)(bb * Hc + h) * Sc + s) * DKc + ch;
#pragma unroll
        for (int q = 0; q < 4; ++q) {
            v8bf o;
#pragma unroll
            for (int e = 0; e < 8; ++e) o[e] = (__bf16)ct[row][ch + q * 8 + e];
            *(v8bf*)(dst + q * 8) = o;
        }
    }
}

// ---- K2: Qrel[b,h,i,r] = Q[b,h,i,:] . rel_k_emb[r,:] ----------------------
__global__ void qrel_kernel(const __bf16* __restrict__ Qb,
                            const float* __restrict__ relk,
                            float* __restrict__ Qrel) {
    int idx = blockIdx.x * blockDim.x + threadIdx.x;
    if (idx >= Bc * Hc * Sc * NRc) return;
    int r   = idx & (NRc - 1);
    int bhi = idx >> 4;
    const __bf16* q  = Qb + (size_t)bhi * DKc;
    const float*  rk = relk + r * DKc;
    float s = 0.f;
#pragma unroll
    for (int d8 = 0; d8 < DKc; d8 += 8) {
        v8bf qv = *(const v8bf*)(q + d8);
#pragma unroll
        for (int d = 0; d < 8; ++d) s += (float)qv[d] * rk[d8 + d];
    }
    Qrel[idx] = s;
}

// ---- K3: fused attention over one (b,h, 16-row strip) ---------------------
__global__ __launch_bounds__(256) void attn_kernel(
    const __bf16* __restrict__ Qb, const __bf16* __restrict__ Kb,
    const __bf16* __restrict__ Vt, const float* __restrict__ Qrel,
    const int* __restrict__ mask, const int* __restrict__ relation,
    const float* __restrict__ rel_v_emb, float* __restrict__ scores_out,
    __bf16* __restrict__ att_bf) {
    __shared__ float sc[16][Sc];        // 64 KB strip (CDNA5: 320 KB LDS/WGP)
    __shared__ float red[16][16];
    __shared__ float rowmax[16];
    __shared__ float rowsum[16];
    __shared__ float srel[16][NRc];
    __shared__ float pacc[16][64];

    int blk = blockIdx.x;
    int it  = blk & 63;
    int bh  = blk >> 6;
    int h   = bh & (Hc - 1);
    int b   = bh >> 4;
    int i0  = it * 16;

    int tid  = threadIdx.x;
    int wave = tid >> 5;
    int lane = tid & 31;
    int col   = lane & 15;
    int rbase = (lane & 16) >> 1;

    srel[tid >> 4][tid & 15] = 0.f;

    const __bf16* Qtile = Qb + ((size_t)bh * Sc + i0) * DKc;
    v16bf aq0 = load_a_bf(Qtile, DKc, lane);
    v16bf aq1 = load_a_bf(Qtile + 32, DKc, lane);

    // ---- pass 1: raw scores = (Q K^T + Qrel[rel]) * 0.125, masked ---------
    {
        const __bf16* K0 = Kb + ((size_t)bh * Sc + wave * 16) * DKc;
        v16bf bk0 = load_b_nmajor(K0, DKc, lane);
        v16bf bk1 = load_b_nmajor(K0 + 32, DKc, lane);
        for (int t = 0; t < 8; ++t) {
            int j0 = (t * 8 + wave) * 16;
            v16bf nk0 = bk0, nk1 = bk1;
            if (t < 7) {
                const __bf16* Kn = Kb + ((size_t)bh * Sc + (j0 + 128)) * DKc;
                nk0 = load_b_nmajor(Kn, DKc, lane);
                nk1 = load_b_nmajor(Kn + 32, DKc, lane);
            }
            v8f acc = {};
            acc = wmma_bf16(aq0, bk0, acc);
            acc = wmma_bf16(aq1, bk1, acc);
#pragma unroll
            for (int e = 0; e < 8; ++e) {
                int il = rbase + e;
                int i  = i0 + il;
                int j  = j0 + col;
                size_t mrIdx = ((size_t)b * Sc + i) * Sc + j;
                int   rel = relation[mrIdx];
                float v = (acc[e] + Qrel[((size_t)bh * Sc + i) * NRc + rel]) * 0.125f;
                if (mask[mrIdx] == 0) v = -1e9f;
                sc[il][j] = v;
            }
            bk0 = nk0; bk1 = nk1;
        }
    }
    __syncthreads();

    int rr = tid >> 4, seg = tid & 15;
    const int c0 = seg * 64;

    // ---- pass 2: row max --------------------------------------------------
    {
        float m = -3.4e38f;
        const float4* srow = (const float4*)&sc[rr][c0];
        for (int c = 0; c < 16; ++c) {
            float4 v = srow[c];
            m = fmaxf(m, fmaxf(fmaxf(v.x, v.y), fmaxf(v.z, v.w)));
        }
        red[rr][seg] = m;
    }
    __syncthreads();
    if (seg == 0) {
        float mm = red[rr][0];
#pragma unroll
        for (int q = 1; q < 16; ++q) mm = fmaxf(mm, red[rr][q]);
        rowmax[rr] = mm;
    }
    __syncthreads();

    // ---- pass 3: exp and row sum ------------------------------------------
    {
        float mx = rowmax[rr];
        float sum = 0.f;
        float4* srow = (float4*)&sc[rr][c0];
        for (int c = 0; c < 16; ++c) {
            float4 v = srow[c];
            v.x = __expf(v.x - mx); v.y = __expf(v.y - mx);
            v.z = __expf(v.z - mx); v.w = __expf(v.w - mx);
            srow[c] = v;
            sum += v.x + v.y + v.z + v.w;
        }
        red[rr][seg] = sum;
    }
    __syncthreads();
    if (seg == 0) {
        float ss = 0.f;
#pragma unroll
        for (int q = 0; q < 16; ++q) ss += red[rr][q];
        rowsum[rr] = ss;
    }
    __syncthreads();

    // ---- pass 4: normalize, write scores, accumulate srel -----------------
    {
        float inv = 1.f / rowsum[rr];
        float4* srow = (float4*)&sc[rr][c0];
        float4* gout = (float4*)(scores_out + ((size_t)bh * Sc + (i0 + rr)) * Sc + c0);
        const int4* rin = (const int4*)(relation + ((size_t)b * Sc + (i0 + rr)) * Sc + c0);
        for (int c = 0; c < 16; ++c) {
            float4 v = srow[c];
            v.x *= inv; v.y *= inv; v.z *= inv; v.w *= inv;
            srow[c] = v;
            gout[c] = v;
            int4 r4 = rin[c];
            atomicAdd(&srel[rr][r4.x], v.x);
            atomicAdd(&srel[rr][r4.y], v.y);
            atomicAdd(&srel[rr][r4.z], v.z);
            atomicAdd(&srel[rr][r4.w], v.w);
        }
    }
    __syncthreads();

    // ---- pass 5: att = P @ V + srel @ rel_v_emb ---------------------------
    v8f acc = {};
    {
        int dt = wave & 3;
        int jh = wave >> 2;
        const __bf16* Vbase = Vt + ((size_t)bh * DKc + dt * 16) * Sc;
        for (int j0 = jh * 512; j0 < jh * 512 + 512; j0 += 32) {
            v16bf ap = load_a_lds_f32(&sc[0][j0], Sc, lane);
            v16bf bv = load_b_nmajor(Vbase + j0, Sc, lane);
            acc = wmma_bf16(ap, bv, acc);
        }
        if (wave >= 4) {
#pragma unroll
            for (int e = 0; e < 8; ++e)
                pacc[rbase + e][dt * 16 + col] = acc[e];
        }
    }
    __syncthreads();
    if (wave < 4) {
#pragma unroll
        for (int e = 0; e < 8; ++e) {
            int il = rbase + e;
            int d  = wave * 16 + col;
            float s = acc[e] + pacc[il][d];
#pragma unroll
            for (int r = 0; r < NRc; ++r) s += srel[il][r] * rel_v_emb[r * DKc + d];
            pacc[il][d] = s;
        }
    }
    __syncthreads();
    {
        int row = tid >> 4, cs = (tid & 15) * 4;
        v4bf o;
#pragma unroll
        for (int q = 0; q < 4; ++q) o[q] = (__bf16)pacc[row][cs + q];
        *(v4bf*)(att_bf + ((size_t)b * Sc + (i0 + row)) * (Hc * DKc) + h * DKc + cs) = o;
    }
}

// ---- K4: FC GEMM  out = att @ fc_w + fc_b (fc_w N-major) ------------------
__global__ __launch_bounds__(256) void gemm_fc_kernel(
    const __bf16* __restrict__ A, const __bf16* __restrict__ WT,
    const float* __restrict__ bias, float* __restrict__ Out) {
    __shared__ __bf16 As[A_LDS_ELEMS];
    __shared__ float ctile[8][16][64];
    int wave = threadIdx.x >> 5;
    int lane = threadIdx.x & 31;
    int mt = blockIdx.x >> 1;
    int nb = ((blockIdx.x & 1) << 3) + wave;
    int m0 = mt * 16, n0 = nb * 64;

    stage_a_panel(As, A + (size_t)m0 * Dc, threadIdx.x);

    v8f acc[4] = {v8f{}, v8f{}, v8f{}, v8f{}};
    gemm_mainloop(As, WT + (size_t)n0 * Dc, lane, acc);

    int col   = lane & 15;
    int rbase = (lane & 16) >> 1;
    float (*ct)[64] = ctile[wave];
#pragma unroll
    for (int t = 0; t < 4; ++t)
#pragma unroll
        for (int e = 0; e < 8; ++e)
            ct[rbase + e][t * 16 + col] = acc[t][e];
    int row = lane & 15, ch = (lane >> 4) * 32;
    int m = m0 + row;
    float* dst = Out + (size_t)m * Dc + n0 + ch;
    const float* bs = bias + n0 + ch;
#pragma unroll
    for (int q = 0; q < 8; ++q) {
        float4 o;
        o.x = ct[row][ch + q * 4 + 0] + bs[q * 4 + 0];
        o.y = ct[row][ch + q * 4 + 1] + bs[q * 4 + 1];
        o.z = ct[row][ch + q * 4 + 2] + bs[q * 4 + 2];
        o.w = ct[row][ch + q * 4 + 3] + bs[q * 4 + 3];
        *(float4*)(dst + q * 4) = o;
    }
}

extern "C" void kernel_launch(void* const* d_in, const int* in_sizes, int n_in,
                              void* d_out, int out_size, void* d_ws, size_t ws_size,
                              hipStream_t stream) {
    const float* query    = (const float*)d_in[0];
    const float* key      = (const float*)d_in[1];
    const float* value    = (const float*)d_in[2];
    const int*   mask     = (const int*)d_in[3];
    const int*   relation = (const int*)d_in[4];
    const float* Wq       = (const float*)d_in[5];
    const float* Wk       = (const float*)d_in[6];
    const float* Wv       = (const float*)d_in[7];
    const float* fc_w     = (const float*)d_in[8];
    const float* fc_b     = (const float*)d_in[9];
    const float* rel_k_emb = (const float*)d_in[10];
    const float* rel_v_emb = (const float*)d_in[11];

    float* out_main   = (float*)d_out;                       // [B,S,D]
    float* out_scores = out_main + (size_t)Bc * Sc * Dc;     // [B,H,S,S]

    const size_t MS = (size_t)Bc * Sc;                       // 2048
    __bf16* p    = (__bf16*)d_ws;
    __bf16* xq   = p; p += MS * Dc;
    __bf16* xk   = p; p += MS * Dc;
    __bf16* xv   = p; p += MS * Dc;
    __bf16* wqT  = p; p += (size_t)Dc * Dc;   // N-major weights
    __bf16* wkT  = p; p += (size_t)Dc * Dc;
    __bf16* wvT  = p; p += (size_t)Dc * Dc;
    __bf16* fcwT = p; p += (size_t)Dc * Dc;
    __bf16* Qb   = p; p += MS * Dc;           // [B,H,S,DK]
    __bf16* Kb   = p; p += MS * Dc;           // [B,H,S,DK]
    __bf16* Vt   = p; p += MS * Dc;           // [B,H,DK,S] (d-major)
    __bf16* attb = p; p += MS * Dc;           // [B,S,H*DK]
    float*  Qrel = (float*)p;                 // [B,H,S,NR]

    const int nAct = Bc * Sc * Dc;
    cvt_bf16_kernel<<<nAct / 4 / 256, 256, 0, stream>>>(query, xq, nAct);
    cvt_bf16_kernel<<<nAct / 4 / 256, 256, 0, stream>>>(key,   xk, nAct);
    cvt_bf16_kernel<<<nAct / 4 / 256, 256, 0, stream>>>(value, xv, nAct);

    dim3 tgrid(Dc / 32, Dc / 32);
    cvtT_bf16_kernel<<<tgrid, 256, 0, stream>>>(Wq,   wqT);
    cvtT_bf16_kernel<<<tgrid, 256, 0, stream>>>(Wk,   wkT);
    cvtT_bf16_kernel<<<tgrid, 256, 0, stream>>>(Wv,   wvT);
    cvtT_bf16_kernel<<<tgrid, 256, 0, stream>>>(fc_w, fcwT);

    // 256 blocks: each = one 16-row m-tile x 512 cols (8 waves)
    gemm_proj_kernel<<<256, 256, 0, stream>>>(xq, wqT, Qb, 0);
    gemm_proj_kernel<<<256, 256, 0, stream>>>(xk, wkT, Kb, 0);
    gemm_proj_kernel<<<256, 256, 0, stream>>>(xv, wvT, Vt, 1);

    qrel_kernel<<<(Bc * Hc * Sc * NRc) / 256, 256, 0, stream>>>(
        Qb, rel_k_emb, Qrel);

    attn_kernel<<<Bc * Hc * (Sc / 16), 256, 0, stream>>>(
        Qb, Kb, Vt, Qrel, mask, relation, rel_v_emb, out_scores, attb);

    gemm_fc_kernel<<<256, 256, 0, stream>>>(attb, fcwT, fc_b, out_main);
}
